// MPNN_14164802142997
// MI455X (gfx1250) — matrix operand order repaired and checked by reading, not compiled, not run on hardware
//
#include <hip/hip_runtime.h>
#include <math.h>

// ---------------- constants from the reference ----------------
#define NATOM   20000
#define NPAIR   800000
#define NWAVE   8
#define NRADIAL 8
#define RMAXL   3
#define PRMAXL  2
#define NORB    9      // RMAXL^2
#define PNORB   4      // PRMAXL^2
#define MP_LOOP 2
#define NSPEC   3
#define NUM_CG  20
#define FEAT    64
#define CSTF    0.2f
#define CUTOFF  5.0f
#define EPSF    1e-8f
#define INV_SQRT8  0.35355339059327373f
#define INV_SQRT2  0.7071067811865475f
#define INV_SQRT6  0.4082482904638631f

// ---------------- WMMA types ----------------
typedef __attribute__((ext_vector_type(16))) _Float16 v16h;
typedef __attribute__((ext_vector_type(8)))  float    v8f;

union AFrag { v16h h; uint4 q[2]; };

__device__ inline float dsilu(float x) { return x / (1.f + __expf(-x)); }

__device__ inline v8f wmma32(const AFrag& a, const AFrag& b, v8f c) {
  return __builtin_amdgcn_wmma_f32_16x16x32_f16(false, a.h, false, b.h,
                                                (short)0, c, false, false);
}

__device__ inline void load_bfrag(AFrag& b, const _Float16* base) {
  b.q[0] = *(const uint4*)(base);
  b.q[1] = *(const uint4*)(base + 16);
}

// silu(f32 tiles) -> f16 LDS tile (16 rows x 64 cols, row-major)
__device__ inline void store_act(_Float16* act, const v8f* hv, int ln, int hi) {
#pragma unroll
  for (int t = 0; t < 4; ++t)
#pragma unroll
    for (int v = 0; v < 8; ++v)
      act[(v + 8 * hi) * 64 + t * 16 + ln] = (_Float16)dsilu(hv[t][v]);
}

// 16x64 (LDS f16) @ 64x64 (WT f16, N-major) -> 4 D tiles
__device__ inline void hidden64(const _Float16* act, const _Float16* __restrict__ WT,
                                int ln, int hi, v8f* dst) {
  AFrag a0, a1, b;
  const _Float16* ar = act + ln * 64;
  a0.q[0] = *(const uint4*)(ar + 8 * hi);
  a0.q[1] = *(const uint4*)(ar + 16 + 8 * hi);
  a1.q[0] = *(const uint4*)(ar + 32 + 8 * hi);
  a1.q[1] = *(const uint4*)(ar + 48 + 8 * hi);
#pragma unroll
  for (int t = 0; t < 4; ++t) {
    v8f c = {};
    const _Float16* wr = WT + (size_t)(t * 16 + ln) * 64 + 8 * hi;
    load_bfrag(b, wr);       c = wmma32(a0, b, c);
    load_bfrag(b, wr + 32);  c = wmma32(a1, b, c);
    dst[t] = c;
  }
}

// ---------------- generic pair-MLP (WMMA): x(32,f16) -> 64 -> 64 -> 64 -> DOUTP ----------------
// out = silu(h + CST*r2) @ wout ; h = x@win ; r1 = silu(h)@w1 ; r2 = silu(r1)@w2
template <int DOUTP>
__global__ void __launch_bounds__(256)
k_mlp_pair(const _Float16* __restrict__ x,     // nrows x 32 (zero padded)
           const _Float16* __restrict__ winT,  // 64 x 32  (out-major, padded)
           const _Float16* __restrict__ w1T,   // 64 x 64
           const _Float16* __restrict__ w2T,   // 64 x 64
           const _Float16* __restrict__ woutT, // DOUTP x 64 (zero padded rows)
           float* __restrict__ out,            // nrows x DOUTP
           int nrows) {
  __shared__ __align__(16) _Float16 lds[8][16 * 64];
  const int wv = threadIdx.x >> 5;
  const int ln = threadIdx.x & 15;
  const int hi = (threadIdx.x >> 4) & 1;
  const int row0 = (blockIdx.x * 8 + wv) * 16;
  if (row0 >= nrows) return;
  _Float16* act = lds[wv];

  // layer 1 (K = 32), A direct from global
  AFrag a, b;
  const _Float16* xr = x + (size_t)(row0 + ln) * 32;
  a.q[0] = *(const uint4*)(xr + 8 * hi);
  a.q[1] = *(const uint4*)(xr + 16 + 8 * hi);
  v8f h[4];
#pragma unroll
  for (int t = 0; t < 4; ++t) {
    v8f c = {};
    load_bfrag(b, winT + (size_t)(t * 16 + ln) * 32 + 8 * hi);
    h[t] = wmma32(a, b, c);
  }
  store_act(act, h, ln, hi);

  v8f r[4];
  hidden64(act, w1T, ln, hi, r);     // r1
  store_act(act, r, ln, hi);
  hidden64(act, w2T, ln, hi, r);     // r2
#pragma unroll
  for (int t = 0; t < 4; ++t)
#pragma unroll
    for (int v = 0; v < 8; ++v)
      h[t][v] = h[t][v] + CSTF * r[t][v];
  store_act(act, h, ln, hi);         // silu(h + CST*r2)

  // output layer (K = 64)
  AFrag a0, a1;
  const _Float16* ar = act + ln * 64;
  a0.q[0] = *(const uint4*)(ar + 8 * hi);
  a0.q[1] = *(const uint4*)(ar + 16 + 8 * hi);
  a1.q[0] = *(const uint4*)(ar + 32 + 8 * hi);
  a1.q[1] = *(const uint4*)(ar + 48 + 8 * hi);
#pragma unroll
  for (int t = 0; t < DOUTP / 16; ++t) {
    v8f c = {};
    const _Float16* wr = woutT + (size_t)(t * 16 + ln) * 64 + 8 * hi;
    load_bfrag(b, wr);       c = wmma32(a0, b, c);
    load_bfrag(b, wr + 32);  c = wmma32(a1, b, c);
#pragma unroll
    for (int v = 0; v < 8; ++v)
      out[(size_t)(row0 + v + 8 * hi) * DOUTP + t * 16 + ln] = c[v];
  }
}

// ---------------- scalar device MLP (tiny row counts / dout==1) ----------------
__device__ void mlp_scalar(const float* x, int din, int dout,
                           const float* win, const float* w1, const float* w2,
                           const float* wout, const float* b, float* out) {
  float h[64], r[64], t[64];
  for (int j = 0; j < 64; ++j) {
    float s = 0.f;
    for (int i = 0; i < din; ++i) s += x[i] * win[i * 64 + j];
    h[j] = s;
  }
  for (int j = 0; j < 64; ++j) t[j] = dsilu(h[j]);
  for (int j = 0; j < 64; ++j) {
    float s = 0.f;
    for (int i = 0; i < 64; ++i) s += t[i] * w1[i * 64 + j];
    r[j] = s;
  }
  for (int j = 0; j < 64; ++j) t[j] = dsilu(r[j]);
  for (int j = 0; j < 64; ++j) {
    float s = 0.f;
    for (int i = 0; i < 64; ++i) s += t[i] * w2[i * 64 + j];
    r[j] = s;
  }
  for (int j = 0; j < 64; ++j) t[j] = dsilu(h[j] + CSTF * r[j]);
  for (int o = 0; o < dout; ++o) {
    float s = b ? b[o] : 0.f;
    for (int i = 0; i < 64; ++i) s += t[i] * wout[i * dout + o];
    out[o] = s;
  }
}

// ---------------- misc kernels ----------------
__global__ void k_zero(float* p, int n) {
  int i = blockIdx.x * blockDim.x + threadIdx.x;
  if (i < n) p[i] = 0.f;
}

__global__ void k_cell(const float* cell, const float* dc, float* symm, float* cellp) {
  int t = threadIdx.x;
  if (t < 9) { int i = t / 3, j = t % 3; symm[t] = 0.5f * (dc[i * 3 + j] + dc[j * 3 + i]); }
  __syncthreads();
  if (t < 9) {
    int i = t / 3, j = t % 3;
    float s = cell[t];
    for (int k = 0; k < 3; ++k) s += cell[i * 3 + k] * symm[k * 3 + j];
    cellp[t] = s;
  }
}

__global__ void k_cfsum(const float* cf, float* acc) {
  int n = blockIdx.x * blockDim.x + threadIdx.x;
  if (n < NATOM) atomicAdd(acc, cf[n]);
}

__global__ void k_atoms(const float* cart, const float* symm, const float* species,
                        const float* rs, float* cartp, int* sidx) {
  int n = blockIdx.x * blockDim.x + threadIdx.x;
  if (n >= NATOM) return;
  for (int d = 0; d < 3; ++d) {
    float s = cart[n * 3 + d];
    for (int k = 0; k < 3; ++k) s += cart[n * 3 + k] * symm[k * 3 + d];
    cartp[n * 3 + d] = s;
  }
  int si = 0;
  for (int c = NSPEC - 1; c >= 0; --c)
    if (fabsf(species[n] - rs[c]) < 0.5f) si = c;
  sidx[n] = si;
}

// 9-row species MLP chain: pair_spec -> emb_coeff / init_ead
__global__ void k_smallmlp(const float* com,
                           const float* ncw, const float* ncw1, const float* ncw2,
                           const float* ncwo, const float* ncb,
                           const float* ngw, const float* ngw1, const float* ngw2,
                           const float* ngwo, const float* ngb,
                           const float* rww, const float* rww1, const float* rww2,
                           const float* rwwo,
                           float* emb, float* iead) {
  int t = threadIdx.x;
  if (t >= 9) return;
  float xs[2] = {com[t * 2], com[t * 2 + 1]};
  float ps[16];
  mlp_scalar(xs, 2, 16, ncw, ncw1, ncw2, ncwo, ncb, ps);
  mlp_scalar(ps, 16, NRADIAL, ngw, ngw1, ngw2, ngwo, ngb, emb + t * NRADIAL);
  mlp_scalar(ps, 16, 16, rww, rww1, rww2, rwwo, nullptr, iead + t * 16);
}

// per-pair geometry + radial-MLP input assembly
__global__ void k_pairs(const float* cartp, const float* cellp, const float* shiftim,
                        const int* nl0, const int* nl1, const float* species,
                        const float* com_spec, const float* emb, const float* iead,
                        float* ave, float* sph, float* eadf, _Float16* x16) {
  int p = blockIdx.x * blockDim.x + threadIdx.x;
  if (p >= NPAIR) return;
  int i0 = nl0[p], i1 = nl1[p];
  float dv[3];
  for (int d = 0; d < 3; ++d) {
    float sh = 0.f;
    for (int k = 0; k < 3; ++k) sh += shiftim[(size_t)p * 3 + k] * cellp[k * 3 + d];
    dv[d] = cartp[i1 * 3 + d] - cartp[i0 * 3 + d] + sh;
  }
  float d2 = dv[0] * dv[0] + dv[1] * dv[1] + dv[2] * dv[2];
  float ngf = (d2 > EPSF) ? 1.f : 0.f;
  float dist = sqrtf(d2 + EPSF);
  float x = dv[0] / dist, y = dv[1] / dist, z = dv[2] / dist;
  float s[9];
  s[0] = 0.28209479177387814f;
  s[1] = 0.4886025119029199f * y;  s[2] = 0.4886025119029199f * z;
  s[3] = 0.4886025119029199f * x;
  s[4] = 1.0925484305920792f * x * y; s[5] = 1.0925484305920792f * y * z;
  s[6] = 0.31539156525252005f * (3.f * z * z - 1.f);
  s[7] = 1.0925484305920792f * x * z;
  s[8] = 0.5462742152960396f * (x * x - y * y);
  float n0 = s[0] * s[0] + EPSF;
  float n1 = s[1] * s[1] + s[2] * s[2] + s[3] * s[3] + EPSF;
  float n2 = s[4] * s[4] + s[5] * s[5] + s[6] * s[6] + s[7] * s[7] + s[8] * s[8] + EPSF;
  s[0] *= 1.f / sqrtf(n0);
  float f1 = 1.7320508075688772f / sqrtf(n1);
  for (int j = 1; j < 4; ++j) s[j] *= f1;
  float f2 = 2.23606797749979f / sqrtf(n2);
  for (int j = 4; j < 9; ++j) s[j] *= f2;
  for (int j = 0; j < 9; ++j) sph[(size_t)p * 9 + j] = s[j];

  float nd = dist / CUTOFF;
  float dp = nd * nd * nd;
  float poly = 1.f - dp * (10.f - 15.f * nd + 6.f * nd * nd);
  float cut = poly * poly * ngf * ((nd < 1.f) ? 1.f : 0.f);
  atomicAdd(ave + i0, cut);

  float s0v = species[i0], s1v = species[i1];
  int pi = 0;
  for (int c = 8; c >= 0; --c)
    if (fabsf(s0v - com_spec[c * 2]) + fabsf(s1v - com_spec[c * 2 + 1]) < 0.5f) pi = c;

  const float* ie = iead + pi * 16;
  const float* em = emb + pi * NRADIAL;
  float* e = eadf + (size_t)p * 32;
  _Float16* xr = x16 + (size_t)p * 32;
  for (int k = 0; k < 8; ++k) e[k] = ie[k] * cut;                 // smooth_ead[:, :8]
  for (int k = 0; k < 8; ++k) xr[k] = (_Float16)(ie[8 + k] * cut); // smooth_ead[:, 8:]
  for (int k = 0; k < 8; ++k) {
    float v = nd * em[k];
    float sc = (fabsf(v) < 1e-6f) ? 1.f
                                  : __sinf(3.14159265358979f * v) / (3.14159265358979f * v);
    xr[8 + k] = (_Float16)(sc * cut * 2.f);
  }
  for (int k = 16; k < 32; ++k) xr[k] = (_Float16)0.f;
}

// f32 weights (din x dout) -> f16 transposed padded (dop x dip)
__global__ void k_wconv(const float* src, int din, int dout, _Float16* dst, int dip, int dop) {
  int id = blockIdx.x * blockDim.x + threadIdx.x;
  if (id >= dip * dop) return;
  int o = id / dip, i = id % dip;
  dst[id] = (o < dout && i < din) ? (_Float16)src[(size_t)i * dout + o] : (_Float16)0.f;
}

// consume wradial (NPAIR x 80): ead cols 8..15, density scatter, co scatter, radial copy
__global__ void k_post_radial(const float* wr, const float* sph, const int* nl0,
                              float* eadf, float* density, float* coraw, float* radc) {
  int p = blockIdx.x * blockDim.x + threadIdx.x;
  if (p >= NPAIR) return;
  const float* r = wr + (size_t)p * 80;
  float* e = eadf + (size_t)p * 32;
  for (int k = 0; k < 8; ++k) e[8 + k] = r[72 + k];
  int a = nl0[p];
  for (int k = 0; k < 8; ++k) atomicAdd(density + (size_t)a * 32 + k, r[64 + k]);
  const float* sp = sph + (size_t)p * 9;
  for (int j = 0; j < 4; ++j) {
    int rr = (j == 0) ? 0 : 1;
    float sj = sp[j];
    for (int k = 0; k < 8; ++k)
      atomicAdd(coraw + (size_t)a * 32 + j * 8 + k, r[rr * 8 + k] * sj);
  }
  float* rc = radc + (size_t)p * 48;
  for (int k = 0; k < 48; ++k) rc[k] = r[16 + k];
}

__global__ void k_co_init(const float* coraw, const float* spec_coeff, const int* sidx,
                          const float* ave, float* co) {
  int id = blockIdx.x * blockDim.x + threadIdx.x;
  if (id >= NATOM * 4) return;
  int n = id >> 2, j = id & 3;
  float avn = ave[n] + EPSF;
  const float* sc = spec_coeff + sidx[n] * 64;
  const float* cr = coraw + (size_t)n * 32 + j * 8;
  float o[8] = {0, 0, 0, 0, 0, 0, 0, 0};
  for (int k = 0; k < 8; ++k) {
    float v = cr[k] / avn;
    for (int m = 0; m < 8; ++m) o[m] += v * sc[k * 8 + m];
  }
  for (int m = 0; m < 8; ++m) co[(size_t)n * 32 + j * 8 + m] = o[m] * INV_SQRT8;
}

__global__ void k_ead12(const float* radc, const float* sph, const float* co,
                        const float* ens, const int* nl0, const int* nl1,
                        float* eadf, int colbase) {
  int p = blockIdx.x * blockDim.x + threadIdx.x;
  if (p >= NPAIR) return;
  const float* rc = radc + (size_t)p * 48;
  const float* sp = sph + (size_t)p * 9;
  const float* c0 = co + (size_t)nl0[p] * 32;
  const float* c1 = co + (size_t)nl1[p] * 32;
  float acc[8] = {0, 0, 0, 0, 0, 0, 0, 0};
  for (int j = 0; j < 4; ++j) {
    int r = (j == 0) ? 0 : 1;
    float sj = sp[j] * ens[j];
    for (int k = 0; k < 8; ++k)
      acc[k] += sj * (rc[r * 8 + k] * c0[j * 8 + k] + rc[16 + r * 8 + k] * c1[j * 8 + k]);
  }
  // norm_corb: /sqrt(PRMAXL); (ead1+ead2)/sqrt(2) -> total 0.5
  for (int k = 0; k < 8; ++k) eadf[(size_t)p * 32 + colbase + k] = acc[k] * 0.5f;
}

__global__ void k_pack(const float* eadf, _Float16* x16, int width) {
  int id = blockIdx.x * blockDim.x + threadIdx.x;
  if (id >= NPAIR * 32) return;
  int c = id & 31;
  x16[id] = (c < width) ? (_Float16)eadf[id] : (_Float16)0.f;
}

__global__ void k_corb(const float* co, const float* contract_it, const int* sidx, float* corb) {
  int id = blockIdx.x * blockDim.x + threadIdx.x;
  if (id >= NATOM * 4) return;
  int n = id >> 2, j = id & 3;
  const float* cc = contract_it + (size_t)sidx[n] * 192;  // a = 0
  const float* cp = co + (size_t)n * 32 + j * 8;
  float o[8] = {0, 0, 0, 0, 0, 0, 0, 0};
  for (int k = 0; k < 8; ++k) {
    float v = cp[k];
    for (int m = 0; m < 8; ++m) o[m] += v * cc[k * 8 + m];
  }
  for (int m = 0; m < 8; ++m) corb[(size_t)n * 32 + j * 8 + m] = o[m] * INV_SQRT8;
}

__global__ void k_scatter_mid(const float* oc, const float* corb, const float* radc,
                              const float* sph, const int* nl0, const int* nl1,
                              const int* index_l,
                              float* iter_orb, float* init_orb, float* sum_orb) {
  int p = blockIdx.x * blockDim.x + threadIdx.x;
  if (p >= NPAIR) return;
  const float* ocp = oc + (size_t)p * 48;
  const float* sp = sph + (size_t)p * 9;
  int a = nl0[p], bn = nl1[p];
  const float* cb = corb + (size_t)bn * 32;
  float* io = iter_orb + (size_t)a * 32;
  for (int j = 0; j < 4; ++j) {
    int r = (j == 0) ? 0 : 1;
    for (int k = 0; k < 8; ++k)
      atomicAdd(io + j * 8 + k, cb[j * 8 + k] * ocp[r * 8 + k]);
  }
  float* ib = init_orb + (size_t)a * 72;
  for (int j = 0; j < 9; ++j) {
    int l = index_l[j];
    float sj = sp[j];
    for (int k = 0; k < 8; ++k)
      atomicAdd(ib + j * 8 + k, ocp[(2 + l) * 8 + k] * sj);
  }
  const float* rc = radc + (size_t)p * 48;
  float* so = sum_orb + (size_t)a * 32;
  for (int j = 0; j < 4; ++j) {
    int r = (j == 0) ? 0 : 1;
    float sj = sp[j];
    for (int k = 0; k < 8; ++k)
      atomicAdd(so + j * 8 + k, rc[32 + r * 8 + k] * sj);
  }
}

__global__ void k_atom_update(const float* iter_orb, const float* init_orb, const float* sum_orb,
                              const float* contract_it, const float* lcoef_it,
                              const int* cg_i1, const int* cg_i2, const int* cg_dst,
                              const float* cg_coef, const float* ens,
                              const int* sidx, const float* ave,
                              float* co, float* density, float* nf_acc, int dcol) {
  int n = blockIdx.x * blockDim.x + threadIdx.x;
  if (n >= NATOM) return;
  float avn = ave[n] + EPSF;
  const float* so = sum_orb + (size_t)n * 32;
  float* cop = co + (size_t)n * 32;
  for (int k = 0; k < 8; ++k) {
    float s = 0.f;
    for (int j = 0; j < 4; ++j) s += so[j * 8 + k] * cop[j * 8 + k] * ens[j];
    density[(size_t)n * 32 + dcol + k] = s * INV_SQRT2;
  }
  float io[72];
  for (int i = 0; i < 72; ++i) io[i] = init_orb[(size_t)n * 72 + i];
  float cgo[32];
  for (int i = 0; i < 32; ++i) cgo[i] = 0.f;
  const float* lc = lcoef_it + (size_t)sidx[n] * NUM_CG * 8;
  for (int c = 0; c < NUM_CG; ++c) {
    int i1 = cg_i1[c], i2 = cg_i2[c], d = cg_dst[c];
    float cf = cg_coef[c];
    for (int k = 0; k < 8; ++k)
      cgo[d * 8 + k] += io[i1 * 8 + k] * io[i2 * 8 + k] * cf * lc[c * 8 + k];
  }
  const float* cc1 = contract_it + (size_t)sidx[n] * 192 + 64;
  const float* cc2 = cc1 + 64;
  const float* it_ = iter_orb + (size_t)n * 32;
  float ss = 0.f, outv[32];
  for (int j = 0; j < 4; ++j)
    for (int m = 0; m < 8; ++m) {
      float s1 = 0.f, s2 = 0.f;
      for (int k = 0; k < 8; ++k) {
        s1 += it_[j * 8 + k] * cc1[k * 8 + m];
        s2 += cgo[j * 8 + k] * cc2[k * 8 + m];
      }
      float v = (s1 + s2) * INV_SQRT8 / avn;
      outv[j * 8 + m] = v;
      ss += v * v;
    }
  for (int i = 0; i < 32; ++i) cop[i] = outv[i];
  atomicAdd(nf_acc, ss);
}

__global__ void k_scale_co(float* co, const float* nf_acc, const float* cfsum) {
  int id = blockIdx.x * blockDim.x + threadIdx.x;
  if (id >= NATOM * 32) return;
  float nf = sqrtf(nf_acc[0] / (cfsum[0] * (float)(PNORB * NWAVE)));
  co[id] /= nf;
}

__global__ void k_scatter_final(const float* radc, const float* sph, const int* nl0,
                                float* sum_orb) {
  int p = blockIdx.x * blockDim.x + threadIdx.x;
  if (p >= NPAIR) return;
  const float* rc = radc + (size_t)p * 48;
  const float* sp = sph + (size_t)p * 9;
  float* so = sum_orb + (size_t)nl0[p] * 32;
  for (int j = 0; j < 4; ++j) {
    int r = (j == 0) ? 0 : 1;
    float sj = sp[j];
    for (int k = 0; k < 8; ++k)
      atomicAdd(so + j * 8 + k,
                (rc[r * 8 + k] + rc[16 + r * 8 + k] + rc[32 + r * 8 + k]) * sj);
  }
}

__global__ void k_density_final(const float* sum_orb, const float* co, const float* ens,
                                float* density) {
  int n = blockIdx.x * blockDim.x + threadIdx.x;
  if (n >= NATOM) return;
  const float* so = sum_orb + (size_t)n * 32;
  const float* cp = co + (size_t)n * 32;
  for (int k = 0; k < 8; ++k) {
    float s = 0.f;
    for (int j = 0; j < 4; ++j) s += so[j * 8 + k] * cp[j * 8 + k] * ens[j];
    density[(size_t)n * 32 + 24 + k] = s * INV_SQRT6;
  }
}

__global__ void k_out(const float* density, const float* ave, const int* sidx,
                      const float* scale, const float* cf,
                      const float* ow, const float* ow1, const float* ow2, const float* owo,
                      float* acc) {
  int n = blockIdx.x * blockDim.x + threadIdx.x;
  if (n >= NATOM) return;
  float xx[32];
  float avn = ave[n] + EPSF;
  for (int i = 0; i < 32; ++i) xx[i] = density[(size_t)n * 32 + i] / avn;
  float o;
  mlp_scalar(xx, 32, 1, ow, ow1, ow2, owo, nullptr, &o);
  int s = sidx[n];
  float a = o * scale[2 * s] + scale[2 * s + 1];
  atomicAdd(acc, a * cf[n]);
}

__global__ void k_final(const float* acc, float* out) {
  if (threadIdx.x == 0 && blockIdx.x == 0) out[0] = acc[0] * 1.0f;  // STD = 1
}

// ---------------- host ----------------
extern "C" void kernel_launch(void* const* d_in, const int* in_sizes, int n_in,
                              void* d_out, int out_size, void* d_ws, size_t ws_size,
                              hipStream_t stream) {
  (void)in_sizes; (void)n_in; (void)out_size; (void)ws_size;
  const float* cart    = (const float*)d_in[0];
  const float* cell    = (const float*)d_in[1];
  const float* dcell   = (const float*)d_in[2];
  const float* shiftim = (const float*)d_in[3];
  const float* cf      = (const float*)d_in[4];
  const float* scale   = (const float*)d_in[5];
  const float* spec_co = (const float*)d_in[6];
  const float* contract= (const float*)d_in[7];
  const float* lcoef   = (const float*)d_in[8];
  const int*   neigh   = (const int*)d_in[47];
  const float* species = (const float*)d_in[48];
  const float* rs      = (const float*)d_in[49];
  const float* com     = (const float*)d_in[50];
  const int*   index_l = (const int*)d_in[51];
  const float* ens     = (const float*)d_in[52];
  const int*   cg_i1   = (const int*)d_in[53];
  const int*   cg_i2   = (const int*)d_in[54];
  const int*   cg_dst  = (const int*)d_in[55];
  const float* cg_cf   = (const float*)d_in[56];
  const int* nl0 = neigh;
  const int* nl1 = neigh + NPAIR;
  float* out = (float*)d_out;

  char* base = (char*)d_ws;
  size_t off = 0;
  auto carve = [&](size_t bytes) -> void* {
    off = (off + 255) & ~(size_t)255;
    void* p = base + off;
    off += bytes;
    return p;
  };

  float* symm    = (float*)carve(9 * 4);
  float* cellp   = (float*)carve(9 * 4);
  float* scal    = (float*)carve(4 * 4);   // [0]=cfsum [1]=nf0 [2]=nf1 [3]=outacc
  float* cartp   = (float*)carve((size_t)NATOM * 3 * 4);
  int*   sidx    = (int*)carve((size_t)NATOM * 4);
  float* ave     = (float*)carve((size_t)NATOM * 4);
  float* emb     = (float*)carve(9 * NRADIAL * 4);
  float* iead    = (float*)carve(9 * 16 * 4);
  float* sph     = (float*)carve((size_t)NPAIR * 9 * 4);
  float* eadf    = (float*)carve((size_t)NPAIR * 32 * 4);
  _Float16* x16  = (_Float16*)carve((size_t)NPAIR * 32 * 2);
  float* big     = (float*)carve((size_t)NPAIR * 80 * 4);  // wradial, reused as orb_coeff
  float* radc    = (float*)carve((size_t)NPAIR * 48 * 4);
  float* coraw   = (float*)carve((size_t)NATOM * 32 * 4);
  float* co      = (float*)carve((size_t)NATOM * 32 * 4);
  float* corb    = (float*)carve((size_t)NATOM * 32 * 4);
  float* iter_orb= (float*)carve((size_t)NATOM * 32 * 4);
  float* init_orb= (float*)carve((size_t)NATOM * 72 * 4);
  float* sum_orb = (float*)carve((size_t)NATOM * 32 * 4);
  float* density = (float*)carve((size_t)NATOM * 32 * 4);

  auto g1 = [](long long n) { return dim3((unsigned)((n + 255) / 256)); };
  dim3 b256(256);

  auto conv = [&](const void* src, int din, int dout, _Float16* dst, int dip, int dop) {
    k_wconv<<<g1((long long)dip * dop), b256, 0, stream>>>((const float*)src, din, dout,
                                                           dst, dip, dop);
  };
  struct WSet { _Float16 *win, *w1, *w2, *wout; };
  auto mkset = [&](int bi, int din, int dout, int doutp) -> WSet {
    WSet s;
    s.win  = (_Float16*)carve(64 * 32 * 2);
    s.w1   = (_Float16*)carve(64 * 64 * 2);
    s.w2   = (_Float16*)carve(64 * 64 * 2);
    s.wout = (_Float16*)carve((size_t)doutp * 64 * 2);
    conv(d_in[bi],     din, 64,   s.win,  32, 64);
    conv(d_in[bi + 1], 64,  64,   s.w1,   64, 64);
    conv(d_in[bi + 2], 64,  64,   s.w2,   64, 64);
    conv(d_in[bi + 3], 64,  dout, s.wout, 64, doutp);
    return s;
  };

  // ---- zero accumulators (every call: deterministic) ----
  k_zero<<<g1(NATOM), b256, 0, stream>>>(ave, NATOM);
  k_zero<<<g1((long long)NATOM * 32), b256, 0, stream>>>(coraw, NATOM * 32);
  k_zero<<<g1((long long)NATOM * 32), b256, 0, stream>>>(density, NATOM * 32);
  k_zero<<<1, 32, 0, stream>>>(scal, 4);

  // ---- prep ----
  k_cell<<<1, 32, 0, stream>>>(cell, dcell, symm, cellp);
  k_cfsum<<<g1(NATOM), b256, 0, stream>>>(cf, scal);
  k_atoms<<<g1(NATOM), b256, 0, stream>>>(cart, symm, species, rs, cartp, sidx);
  k_smallmlp<<<1, 16, 0, stream>>>(com,
      (const float*)d_in[9],  (const float*)d_in[10], (const float*)d_in[11],
      (const float*)d_in[12], (const float*)d_in[13],
      (const float*)d_in[14], (const float*)d_in[15], (const float*)d_in[16],
      (const float*)d_in[17], (const float*)d_in[18],
      (const float*)d_in[19], (const float*)d_in[20], (const float*)d_in[21],
      (const float*)d_in[22],
      emb, iead);
  k_pairs<<<g1(NPAIR), b256, 0, stream>>>(cartp, cellp, shiftim, nl0, nl1, species,
                                          com, emb, iead, ave, sph, eadf, x16);

  // ---- f16 weight prep ----
  WSet wrad = mkset(23, 16, 80, 80);
  WSet wmp0 = mkset(27, 24, 40, 48);
  WSet wmp1 = mkset(31, 32, 40, 48);
  WSet wed0 = mkset(35, 24, 48, 48);
  WSet wed1 = mkset(39, 32, 48, 48);
  WSet wmp[2] = {wmp0, wmp1};
  WSet wed[2] = {wed0, wed1};

  const unsigned mlpBlocks = NPAIR / 128;  // 8 waves x 16 rows per block

  // ---- radial MLP (WMMA) + init stage ----
  k_mlp_pair<80><<<mlpBlocks, b256, 0, stream>>>(x16, wrad.win, wrad.w1, wrad.w2, wrad.wout,
                                                 big, NPAIR);
  k_post_radial<<<g1(NPAIR), b256, 0, stream>>>(big, sph, nl0, eadf, density, coraw, radc);
  k_co_init<<<g1((long long)NATOM * 4), b256, 0, stream>>>(coraw, spec_co, sidx, ave, co);

  // ---- MP loops ----
  for (int it = 0; it < MP_LOOP; ++it) {
    const float* cit = contract + (size_t)it * NSPEC * 3 * 64;
    const float* lit = lcoef + (size_t)it * NSPEC * NUM_CG * 8;
    k_ead12<<<g1(NPAIR), b256, 0, stream>>>(radc, sph, co, ens, nl0, nl1, eadf, 16 + 8 * it);
    k_pack<<<g1((long long)NPAIR * 32), b256, 0, stream>>>(eadf, x16, 24 + 8 * it);
    k_mlp_pair<48><<<mlpBlocks, b256, 0, stream>>>(x16, wmp[it].win, wmp[it].w1, wmp[it].w2,
                                                   wmp[it].wout, big, NPAIR);  // orb_coeff
    k_zero<<<g1((long long)NATOM * 32), b256, 0, stream>>>(iter_orb, NATOM * 32);
    k_zero<<<g1((long long)NATOM * 72), b256, 0, stream>>>(init_orb, NATOM * 72);
    k_zero<<<g1((long long)NATOM * 32), b256, 0, stream>>>(sum_orb, NATOM * 32);
    k_corb<<<g1((long long)NATOM * 4), b256, 0, stream>>>(co, cit, sidx, corb);
    k_scatter_mid<<<g1(NPAIR), b256, 0, stream>>>(big, corb, radc, sph, nl0, nl1, index_l,
                                                  iter_orb, init_orb, sum_orb);
    k_atom_update<<<g1(NATOM), b256, 0, stream>>>(iter_orb, init_orb, sum_orb, cit, lit,
                                                  cg_i1, cg_i2, cg_dst, cg_cf, ens,
                                                  sidx, ave, co, density, scal + 1 + it,
                                                  8 + 8 * it);
    k_scale_co<<<g1((long long)NATOM * 32), b256, 0, stream>>>(co, scal + 1 + it, scal);
    k_mlp_pair<48><<<mlpBlocks, b256, 0, stream>>>(x16, wed[it].win, wed[it].w1, wed[it].w2,
                                                   wed[it].wout, radc, NPAIR);  // new radial
  }

  // ---- final density block + output MLP ----
  k_zero<<<g1((long long)NATOM * 32), b256, 0, stream>>>(sum_orb, NATOM * 32);
  k_scatter_final<<<g1(NPAIR), b256, 0, stream>>>(radc, sph, nl0, sum_orb);
  k_density_final<<<g1(NATOM), b256, 0, stream>>>(sum_orb, co, ens, density);
  k_out<<<g1(NATOM), b256, 0, stream>>>(density, ave, sidx, scale, cf,
                                        (const float*)d_in[43], (const float*)d_in[44],
                                        (const float*)d_in[45], (const float*)d_in[46],
                                        scal + 3);
  k_final<<<1, 32, 0, stream>>>(scal + 3, out);
}
